// MultiHeadedGlobalAttention_13176959664319
// MI455X (gfx1250) — matrix operand order repaired and checked by reading, not compiled, not run on hardware
//
#include <hip/hip_runtime.h>
#include <cstdint>
#include <cstddef>

// ---------------------------------------------------------------------------
// MultiHeadedGlobalAttention for MI455X (gfx1250, wave32, WMMA)
//
// Shapes (fixed by reference): B=4, L=S=2048, E=512, H=8, dh=64, G=16.
//
// Streaming reformulation: resp_* are ratios of prefix sums of
// (w, w*K, w*V) with w = exp(cs . K). 3-pass blocked scan, then the five
// 512x512 GEMMs run through v_wmma_f32_16x16x32_bf16 with fp32 accumulate.
// ---------------------------------------------------------------------------

typedef __bf16   bf16x16 __attribute__((ext_vector_type(16)));
typedef float    f32x8   __attribute__((ext_vector_type(8)));
typedef float    f32x4   __attribute__((ext_vector_type(4)));
typedef unsigned u32x4   __attribute__((ext_vector_type(4)));

constexpr int B_   = 4;
constexpr int L_   = 2048;
constexpr int E_   = 512;      // = KD = VD
constexpr int H_   = 8;
constexpr int DH   = 64;       // per-head dim
constexpr int G_   = 16;       // global queries
constexpr int CHUNK = 64;      // scan chunk length
constexpr int NCH   = L_ / CHUNK;          // 32 chunks
constexpr int STATE = 16 + 2 * G_ * DH;    // 2064 floats: n[16], A[16][64], Bv[16][64]

__device__ __forceinline__ unsigned short f2bf(float x) {
    unsigned u = __float_as_uint(x);
    u += 0x7fffu + ((u >> 16) & 1u);          // round-to-nearest-even
    return (unsigned short)(u >> 16);
}
__device__ __forceinline__ unsigned packbf(float lo, float hi) {
    return (unsigned)f2bf(lo) | ((unsigned)f2bf(hi) << 16);
}

// --------------------------------------------------------------------------
// Weight prep: fp32 W[k][n] (512x512) -> bf16 Wt[n][k] (transposed so each
// WMMA B-lane reads its k values as two 16-byte runs from LDS).
// --------------------------------------------------------------------------
__global__ __launch_bounds__(256) void convert_w(const float* __restrict__ W,
                                                 unsigned short* __restrict__ Wt) {
    int idx = blockIdx.x * 256 + threadIdx.x;   // 0 .. 512*512-1
    int k = idx >> 9, n = idx & 511;
    Wt[n * 512 + k] = f2bf(W[(size_t)k * 512 + n]);
}

// --------------------------------------------------------------------------
// C[M x 512] = A[M x 512] (fp32) * W (bf16, stored transposed [n][k])
// 128x128x32 tiles, 256 threads = 8 waves (4x2 wave grid), each wave does a
// 32x64 C patch = 2x4 accumulators of v_wmma_f32_16x16x32_bf16.
// --------------------------------------------------------------------------
union Frag16 { bf16x16 v; u32x4 q[2]; };

__global__ __launch_bounds__(256) void gemm_bf16(const float* __restrict__ A,
                                                 const unsigned short* __restrict__ Wt,
                                                 float* __restrict__ Cout, int M) {
    constexpr int K   = 512;
    constexpr int LDA = 40;   // padded row stride (ushorts): 80B, 16B-aligned chunks
    __shared__ alignas(16) unsigned short sA[128 * LDA];
    __shared__ alignas(16) unsigned short sB[128 * LDA];

    const int t    = threadIdx.x;
    const int lane = t & 31;
    const int wave = t >> 5;
    const int wm   = wave & 3;            // 4 waves along M (32 rows each)
    const int wn   = wave >> 2;           // 2 waves along N (64 cols each)
    const int row0 = blockIdx.x * 128;
    const int n0   = blockIdx.y * 128;

    f32x8 acc[2][4];
    f32x8 zacc = {};
#pragma unroll
    for (int i = 0; i < 2; ++i)
#pragma unroll
        for (int j = 0; j < 4; ++j) acc[i][j] = zacc;

    const int ar = t >> 1;                // staging row 0..127
    const int ah = t & 1;                 // k half (16 elements)
    const int hf  = lane >> 4;
    const int l15 = lane & 15;

    for (int k0 = 0; k0 < K; k0 += 32) {
        if (k0) __syncthreads();
        // ---- stage A tile: fp32 -> bf16 into LDS -------------------------
        const int grow = row0 + ar;
        const float* ap = A + (size_t)grow * K + k0 + ah * 16;
        f32x4 f0, f1, f2, f3;
        if (grow < M) {
            f0 = *(const f32x4*)(ap + 0);
            f1 = *(const f32x4*)(ap + 4);
            f2 = *(const f32x4*)(ap + 8);
            f3 = *(const f32x4*)(ap + 12);
        } else {
            f32x4 z = {0.f, 0.f, 0.f, 0.f};
            f0 = f1 = f2 = f3 = z;
        }
        u32x4 p0, p1;
        p0[0] = packbf(f0.x, f0.y); p0[1] = packbf(f0.z, f0.w);
        p0[2] = packbf(f1.x, f1.y); p0[3] = packbf(f1.z, f1.w);
        p1[0] = packbf(f2.x, f2.y); p1[1] = packbf(f2.z, f2.w);
        p1[2] = packbf(f3.x, f3.y); p1[3] = packbf(f3.z, f3.w);
        *(u32x4*)&sA[ar * LDA + ah * 16]     = p0;
        *(u32x4*)&sA[ar * LDA + ah * 16 + 8] = p1;
        // ---- stage B tile: bf16 transposed weight [n][k] -----------------
        const unsigned short* wp = Wt + (size_t)(n0 + ar) * K + k0 + ah * 16;
        u32x4 w0 = *(const u32x4*)(wp + 0);
        u32x4 w1 = *(const u32x4*)(wp + 8);
        *(u32x4*)&sB[ar * LDA + ah * 16]     = w0;
        *(u32x4*)&sB[ar * LDA + ah * 16 + 8] = w1;
        if (k0 + 32 < K) {
            if (grow < M) __builtin_prefetch(ap + 32, 0, 1);   // global_prefetch_b8
            __builtin_prefetch(wp + 32, 0, 1);
        }
        __syncthreads();
        // ---- WMMA: operand layouts per CDNA5 ISA 7.12.2 ------------------
        Frag16 fa[2], fb[4];
#pragma unroll
        for (int mt = 0; mt < 2; ++mt) {
            int m = wm * 32 + mt * 16 + l15;
            fa[mt].q[0] = *(const u32x4*)&sA[m * LDA + hf * 8];        // K 0..7 / 8..15
            fa[mt].q[1] = *(const u32x4*)&sA[m * LDA + 16 + hf * 8];   // K 16..23 / 24..31
        }
#pragma unroll
        for (int nt = 0; nt < 4; ++nt) {
            int n = wn * 64 + nt * 16 + l15;
            fb[nt].q[0] = *(const u32x4*)&sB[n * LDA + hf * 8];        // K 0..7 / 8..15
            fb[nt].q[1] = *(const u32x4*)&sB[n * LDA + 16 + hf * 8];   // K 16..23 / 24..31
        }
#pragma unroll
        for (int mt = 0; mt < 2; ++mt)
#pragma unroll
            for (int nt = 0; nt < 4; ++nt)
                acc[mt][nt] = __builtin_amdgcn_wmma_f32_16x16x32_bf16(
                    false, fa[mt].v, false, fb[nt].v, (short)0, acc[mt][nt],
                    false, false);
    }
    // ---- epilogue: C layout VGPR r -> (M = r + half*8, N = lane&15) ------
#pragma unroll
    for (int mt = 0; mt < 2; ++mt)
#pragma unroll
        for (int nt = 0; nt < 4; ++nt)
#pragma unroll
            for (int r = 0; r < 8; ++r) {
                int row = row0 + wm * 32 + mt * 16 + hf * 8 + r;
                int col = n0 + wn * 64 + nt * 16 + l15;
                if (row < M) Cout[(size_t)row * 512 + col] = acc[mt][nt][r];
            }
}

// --------------------------------------------------------------------------
// Pass 1: per (b,h,chunk) totals of (w, w*K, w*V).
// Thread t = (g = t>>4, ci = t&15) owns 4 of the 64 head dims for group g.
// w reduction is a 16-lane butterfly (wave32: lanes [0..15]/[16..31]).
// --------------------------------------------------------------------------
__global__ __launch_bounds__(256) void chunk_sums(const float* __restrict__ ks,
                                                  const float* __restrict__ vs,
                                                  const float* __restrict__ cs,
                                                  float* __restrict__ cstate) {
    __shared__ alignas(16) float kt[CHUNK * DH];
    __shared__ alignas(16) float vt[CHUNK * DH];
    const int bid = blockIdx.x;
    const int c   = bid & (NCH - 1);
    const int bh  = bid / NCH;
    const int b   = bh / H_, h = bh % H_;
    const int t   = threadIdx.x;
    const int g   = t >> 4, ci = t & 15;

    const float* kbase = ks + ((size_t)b * L_ + c * CHUNK) * 512 + h * DH;
    const float* vbase = vs + ((size_t)b * L_ + c * CHUNK) * 512 + h * DH;
#pragma unroll
    for (int i = 0; i < 4; ++i) {
        int fi = t + i * 256;                 // float4 index 0..1023
        int s = fi >> 4, j = fi & 15;
        *(f32x4*)&kt[s * DH + j * 4] = *(const f32x4*)(kbase + (size_t)s * 512 + j * 4);
        *(f32x4*)&vt[s * DH + j * 4] = *(const f32x4*)(vbase + (size_t)s * 512 + j * 4);
    }
    float csv[4];
    const float* cp = cs + ((size_t)b * G_ + g) * 512 + h * DH + ci * 4;
#pragma unroll
    for (int j = 0; j < 4; ++j) csv[j] = cp[j];
    __syncthreads();

    float sw = 0.f, sA[4] = {0.f, 0.f, 0.f, 0.f}, sV[4] = {0.f, 0.f, 0.f, 0.f};
    for (int s = 0; s < CHUNK; ++s) {
        const float* kr = &kt[s * DH + ci * 4];
        const float* vr = &vt[s * DH + ci * 4];
        float d = csv[0] * kr[0] + csv[1] * kr[1] + csv[2] * kr[2] + csv[3] * kr[3];
#pragma unroll
        for (int m = 1; m < 16; m <<= 1) d += __shfl_xor(d, m, 16);
        float w = __expf(d);
        sw += w;
#pragma unroll
        for (int j = 0; j < 4; ++j) { sA[j] += w * kr[j]; sV[j] += w * vr[j]; }
    }
    float* st = cstate + (size_t)bid * STATE;
    if (ci == 0) st[g] = sw;
#pragma unroll
    for (int j = 0; j < 4; ++j) {
        st[16 + g * DH + ci * 4 + j]        = sA[j];
        st[16 + 1024 + g * DH + ci * 4 + j] = sV[j];
    }
}

// --------------------------------------------------------------------------
// Pass 2: per (b,h), in-place exclusive prefix over the 32 chunk states.
// --------------------------------------------------------------------------
__global__ __launch_bounds__(256) void chunk_prefix(float* __restrict__ cstate) {
    const int bh = blockIdx.x, t = threadIdx.x;
    float run[9];
#pragma unroll
    for (int i = 0; i < 9; ++i) run[i] = 0.f;
    for (int c = 0; c < NCH; ++c) {
        float* st = cstate + ((size_t)bh * NCH + c) * STATE;
#pragma unroll
        for (int i = 0; i < 9; ++i) {
            int idx = t + i * 256;
            if (idx < STATE) {
                float v = st[idx];
                st[idx] = run[i];     // exclusive prefix for this chunk
                run[i] += v;
            }
        }
    }
}

// --------------------------------------------------------------------------
// Pass 3: inclusive within-chunk scan + logits + 16-way softmax + output.
// --------------------------------------------------------------------------
__global__ __launch_bounds__(256) void attn_scan(const float* __restrict__ qs,
                                                 const float* __restrict__ ks,
                                                 const float* __restrict__ vs,
                                                 const float* __restrict__ cs,
                                                 const float* __restrict__ cstate,
                                                 float* __restrict__ aout) {
    __shared__ alignas(16) float kt[CHUNK * DH];
    __shared__ alignas(16) float vt[CHUNK * DH];
    __shared__ alignas(16) float qt[CHUNK * DH];
    __shared__ float s_logit[G_];
    __shared__ float s_acc[G_ * DH];

    const int bid = blockIdx.x;
    const int c   = bid & (NCH - 1);
    const int bh  = bid / NCH;
    const int b   = bh / H_, h = bh % H_;
    const int t   = threadIdx.x;
    const int g   = t >> 4, ci = t & 15;

    const size_t rowbase = ((size_t)b * L_ + c * CHUNK) * 512 + h * DH;
    const float* kbase = ks + rowbase;
    const float* vbase = vs + rowbase;
    const float* qbase = qs + rowbase;
#pragma unroll
    for (int i = 0; i < 4; ++i) {
        int fi = t + i * 256;
        int s = fi >> 4, j = fi & 15;
        *(f32x4*)&kt[s * DH + j * 4] = *(const f32x4*)(kbase + (size_t)s * 512 + j * 4);
        *(f32x4*)&vt[s * DH + j * 4] = *(const f32x4*)(vbase + (size_t)s * 512 + j * 4);
        *(f32x4*)&qt[s * DH + j * 4] = *(const f32x4*)(qbase + (size_t)s * 512 + j * 4);
    }
    float csv[4];
    const float* cp = cs + ((size_t)b * G_ + g) * 512 + h * DH + ci * 4;
#pragma unroll
    for (int j = 0; j < 4; ++j) csv[j] = cp[j];

    // exclusive-prefix initial state
    const float* st = cstate + (size_t)bid * STATE;
    float n = st[g];
    float Areg[4], Vreg[4];
#pragma unroll
    for (int j = 0; j < 4; ++j) {
        Areg[j] = st[16 + g * DH + ci * 4 + j];
        Vreg[j] = st[16 + 1024 + g * DH + ci * 4 + j];
    }
    __syncthreads();

    for (int s = 0; s < CHUNK; ++s) {
        const float* kr = &kt[s * DH + ci * 4];
        const float* vr = &vt[s * DH + ci * 4];
        const float* qr = &qt[s * DH + ci * 4];
        float d = csv[0] * kr[0] + csv[1] * kr[1] + csv[2] * kr[2] + csv[3] * kr[3];
#pragma unroll
        for (int m = 1; m < 16; m <<= 1) d += __shfl_xor(d, m, 16);
        float w = __expf(d);
        n += w;                                   // inclusive, like reference cumsum
#pragma unroll
        for (int j = 0; j < 4; ++j) { Areg[j] += w * kr[j]; Vreg[j] += w * vr[j]; }

        float qd = qr[0] * Areg[0] + qr[1] * Areg[1] + qr[2] * Areg[2] + qr[3] * Areg[3];
#pragma unroll
        for (int m = 1; m < 16; m <<= 1) qd += __shfl_xor(qd, m, 16);
        if (ci == 0) s_logit[g] = qd / n;         // logit = q . (A_cum / n)
        __syncthreads();

        float mx = s_logit[0];
#pragma unroll
        for (int gg = 1; gg < G_; ++gg) mx = fmaxf(mx, s_logit[gg]);
        float den = 0.f;
#pragma unroll
        for (int gg = 0; gg < G_; ++gg) den += __expf(s_logit[gg] - mx);
        float p = __expf(s_logit[g] - mx) / den;
        float r = p / n;                          // softmax weight over resp_v = Bv/n
#pragma unroll
        for (int j = 0; j < 4; ++j) s_acc[g * DH + ci * 4 + j] = r * Vreg[j];
        __syncthreads();

        if (t < DH) {
            float o = 0.f;
#pragma unroll
            for (int gg = 0; gg < G_; ++gg) o += s_acc[gg * DH + t];
            aout[((size_t)b * L_ + c * CHUNK + s) * 512 + h * DH + t] = o;
        }
        __syncthreads();
    }
}

// --------------------------------------------------------------------------
// Host side.
// Workspace layout (bytes): 5 bf16 weights | qs | ks | vs | cs | attn | cstate
// --------------------------------------------------------------------------
extern "C" void kernel_launch(void* const* d_in, const int* in_sizes, int n_in,
                              void* d_out, int out_size, void* d_ws, size_t ws_size,
                              hipStream_t stream) {
    (void)in_sizes; (void)n_in; (void)out_size; (void)ws_size;
    const float* q   = (const float*)d_in[0];
    const float* k   = (const float*)d_in[1];
    const float* v   = (const float*)d_in[2];
    const float* ctx = (const float*)d_in[3];
    const float* W[5] = { (const float*)d_in[4],   // query_proj
                          (const float*)d_in[5],   // key_proj
                          (const float*)d_in[6],   // value_proj
                          (const float*)d_in[7],   // output_proj
                          (const float*)d_in[8] }; // context_proj
    float* out = (float*)d_out;

    constexpr size_t SZ_W   = (size_t)512 * 512 * sizeof(unsigned short); // 512 KB
    constexpr size_t SZ_ACT = (size_t)B_ * L_ * 512 * sizeof(float);      // 16.78 MB
    constexpr size_t SZ_CS  = (size_t)B_ * G_ * 512 * sizeof(float);      // 128 KB

    char* p = (char*)d_ws;
    unsigned short* wbf[5];
    for (int i = 0; i < 5; ++i) { wbf[i] = (unsigned short*)p; p += SZ_W; }
    float* qsb = (float*)p; p += SZ_ACT;
    float* ksb = (float*)p; p += SZ_ACT;
    float* vsb = (float*)p; p += SZ_ACT;
    float* csb = (float*)p; p += SZ_CS;
    float* ab  = (float*)p; p += SZ_ACT;
    float* cst = (float*)p;  // B_*H_*NCH*STATE floats = 8.45 MB

    dim3 blk(256);
    for (int i = 0; i < 5; ++i)
        convert_w<<<1024, blk, 0, stream>>>(W[i], wbf[i]);

    dim3 gact(64, 4);   // 8192/128 x 512/128
    gemm_bf16<<<gact, blk, 0, stream>>>(q, wbf[0], qsb, B_ * L_);
    gemm_bf16<<<gact, blk, 0, stream>>>(k, wbf[1], ksb, B_ * L_);
    gemm_bf16<<<gact, blk, 0, stream>>>(v, wbf[2], vsb, B_ * L_);
    gemm_bf16<<<dim3(1, 4), blk, 0, stream>>>(ctx, wbf[4], csb, B_ * G_);

    chunk_sums <<<B_ * H_ * NCH, blk, 0, stream>>>(ksb, vsb, csb, cst);
    chunk_prefix<<<B_ * H_,      blk, 0, stream>>>(cst);
    attn_scan  <<<B_ * H_ * NCH, blk, 0, stream>>>(qsb, ksb, vsb, csb, cst, ab);

    gemm_bf16<<<gact, blk, 0, stream>>>(ab, wbf[3], out, B_ * L_);
}
